// JKNet_5050881540195
// MI455X (gfx1250) — matrix operand order repaired and checked by reading, not compiled, not run on hardware
//
#include <hip/hip_runtime.h>
#include <hip/hip_bf16.h>

// ---------------- problem constants (from reference) ----------------
#define NN   170000
#define EE   2720000
#define IND  128
#define HD   256
#define LL   4
#define OUTD 40
#define BN_EPS 1e-5f

// ---------------- types ----------------
typedef __bf16 v16bf __attribute__((ext_vector_type(16)));
typedef float  v8f   __attribute__((ext_vector_type(8)));
typedef int    v4i   __attribute__((vector_size(16)));   // matches builtin param type

#define AS_GLOBAL __attribute__((address_space(1)))
#define AS_LOCAL  __attribute__((address_space(3)))

__device__ __forceinline__ unsigned short f32_to_bf16_rne(float f) {
    unsigned int u = __float_as_uint(f);
    unsigned int r = 0x7FFFu + ((u >> 16) & 1u);
    return (unsigned short)((u + r) >> 16);
}

// ---- gfx1250 async global->LDS copy (ASYNCcnt path) ----
__device__ __forceinline__ void async_copy_g2l_b128(const void* g, void* l) {
    void* gnc = const_cast<void*>(g);
#if __has_builtin(__builtin_amdgcn_global_load_async_to_lds_b128)
    __builtin_amdgcn_global_load_async_to_lds_b128(
        (AS_GLOBAL v4i*)gnc, (AS_LOCAL v4i*)l, 0, 0);
#else
    unsigned loff = (unsigned)(size_t)(AS_LOCAL void*)l;
    asm volatile("global_load_async_to_lds_b128 %0, %1, off"
                 :: "v"(loff), "v"((unsigned long long)(size_t)gnc)
                 : "memory");
#endif
}

__device__ __forceinline__ void wait_async0() {
#if __has_builtin(__builtin_amdgcn_s_wait_asynccnt)
    __builtin_amdgcn_s_wait_asynccnt(0);
#else
    asm volatile("s_wait_asynccnt 0" ::: "memory");
#endif
}

// ---------------- utility kernels ----------------
__global__ __launch_bounds__(256) void k_zero_f32(float* p, long long n) {
    long long i = (long long)blockIdx.x * blockDim.x + threadIdx.x;
    if (i < n) p[i] = 0.0f;
}

__global__ __launch_bounds__(256) void k_f32_to_bf16(const float* __restrict__ src,
                                                     unsigned short* __restrict__ dst,
                                                     long long n) {
    long long i = (long long)blockIdx.x * blockDim.x + threadIdx.x;
    if (i < n) dst[i] = f32_to_bf16_rne(src[i]);
}

// ---------------- degree / normalization ----------------
__global__ __launch_bounds__(256) void k_edge_deg(const int* __restrict__ dst,
                                                  float* __restrict__ deg, int E) {
    int e = blockIdx.x * blockDim.x + threadIdx.x;
    if (e < E) atomicAdd(&deg[dst[e]], 1.0f);
}

__global__ __launch_bounds__(256) void k_node_norm(const float* __restrict__ deg,
                                                   float* __restrict__ dinv,
                                                   float* __restrict__ invd, int n) {
    int i = blockIdx.x * blockDim.x + threadIdx.x;
    if (i < n) {
        float d = deg[i] + 1.0f;      // + self loop, PyG gcn_norm style
        dinv[i] = rsqrtf(d);
        invd[i] = 1.0f / d;
    }
}

// ---------------- edge scatter-add (memory-bound core) ----------------
// one wave32 per edge; 256 floats per row via float4, atomic f32 scatter
__global__ __launch_bounds__(256) void k_aggregate(const float* __restrict__ hW,
                                                   const int* __restrict__ src,
                                                   const int* __restrict__ dst,
                                                   const float* __restrict__ dinv,
                                                   float* __restrict__ agg, int E) {
    int e = blockIdx.x * 8 + (threadIdx.x >> 5);
    if (e >= E) return;
    int lane = threadIdx.x & 31;
    int s = src[e], d = dst[e];
    float w = dinv[s] * dinv[d];
    const float4* hs = (const float4*)(hW + (size_t)s * HD);
    float* ad = agg + (size_t)d * HD;
#pragma unroll
    for (int j = 0; j < 2; ++j) {
        int i4 = lane + j * 32;            // 64 float4 per 256-wide row
        float4 v = hs[i4];
        atomicAdd(ad + i4 * 4 + 0, v.x * w);
        atomicAdd(ad + i4 * 4 + 1, v.y * w);
        atomicAdd(ad + i4 * 4 + 2, v.z * w);
        atomicAdd(ad + i4 * 4 + 3, v.w * w);
    }
}

// ---------------- self-loop + bias + BN statistics ----------------
__global__ __launch_bounds__(256) void k_finish_stats(float* __restrict__ hpre,          // agg, in/out
                                                      const float* __restrict__ hW,
                                                      const float* __restrict__ invd,
                                                      const float* __restrict__ bias,
                                                      float* __restrict__ bsum,
                                                      float* __restrict__ bsq, int rows) {
    int c = threadIdx.x;
    int r0 = blockIdx.x * 64;
    int r1 = r0 + 64; if (r1 > rows) r1 = rows;
    float bc = bias[c];
    float s = 0.f, q = 0.f;
    for (int r = r0; r < r1; ++r) {
        size_t idx = (size_t)r * HD + c;
        float v = hpre[idx] + hW[idx] * invd[r] + bc;
        hpre[idx] = v;
        s += v; q += v * v;
    }
    atomicAdd(&bsum[c], s);
    atomicAdd(&bsq[c], q);
}

// ---------------- BN(train stats) + ReLU + bf16 pack into JK concat ----------------
__global__ __launch_bounds__(256) void k_bn_relu_pack(const float* __restrict__ hpre,
                                                      const float* __restrict__ bsum,
                                                      const float* __restrict__ bsq,
                                                      const float* __restrict__ gamma,
                                                      const float* __restrict__ beta,
                                                      unsigned short* __restrict__ out, // concat slice, stride 1024
                                                      int rows, float invN) {
    long long i = (long long)blockIdx.x * blockDim.x + threadIdx.x;
    if (i >= (long long)rows * HD) return;
    int r = (int)(i >> 8), c = (int)(i & 255);
    float mean = bsum[c] * invN;
    float var  = bsq[c] * invN - mean * mean;
    float sc   = gamma[c] * rsqrtf(var + BN_EPS);
    float v = (hpre[i] - mean) * sc + beta[c];
    v = fmaxf(v, 0.0f);
    out[(size_t)r * (LL * HD) + c] = f32_to_bf16_rne(v);
}

// ---------------- WMMA bf16 GEMM:  C = A(bf16)[MxK] * B(bf16)[KxNcols] (+bias) ----------------
// block tile 128x64, 8 waves, wave tile 32x32 (2x2 v_wmma_f32_16x16x32_bf16), K step 32.
// A tiles staged via async global->LDS b128 copies, double-buffered; B staged via
// coalesced uint4 loads + transposed b16 LDS stores ([n][k] so K-pairs are 32-bit).
#define BM 128
#define BN 64
#define BK 32

__global__ __launch_bounds__(256) void k_gemm_bf16_wmma(const unsigned short* __restrict__ A, int lda,
                                                        const unsigned short* __restrict__ B, int ldb,
                                                        const float* __restrict__ bias,
                                                        float* __restrict__ C, int ldc,
                                                        int M, int K, int Ncols) {
    __shared__ __align__(16) unsigned short sA[2][BM * BK];   // [row][k]
    __shared__ __align__(16) unsigned short sB[2][BN * BK];   // [n][k]

    const int tid  = threadIdx.x;
    const int bm   = blockIdx.x * BM;
    const int bn   = blockIdx.y * BN;
    const int wave = tid >> 5;
    const int lane = tid & 31;
    const int half = lane >> 4;       // 0: lanes 0-15, 1: lanes 16-31
    const int l16  = lane & 15;
    const int waveM = (wave & 3) * 32;
    const int waveN = (wave >> 2) * 32;

    // per-thread A-tile chunk coordinates (2 x b128 chunks of the 128x32 tile)
    int arow[2], akp[2];
    size_t abase[2];
#pragma unroll
    for (int cc = 0; cc < 2; ++cc) {
        int c = tid + cc * 256;
        arow[cc] = c >> 2;
        akp[cc]  = (c & 3) * 8;
        int gr = bm + arow[cc]; if (gr >= M) gr = M - 1;   // clamp: never stored
        abase[cc] = (size_t)gr * lda + akp[cc];
    }
    // per-thread B-tile chunk coordinates (1 x uint4 of the 32x64 tile)
    const int bk  = tid >> 3;
    const int bn8 = (tid & 7) * 8;
    const bool bvalid = (bn + bn8) < Ncols;                 // Ncols % 8 == 0 here

    auto issueA = [&](int k0, int buf) {
#pragma unroll
        for (int cc = 0; cc < 2; ++cc)
            async_copy_g2l_b128(A + abase[cc] + k0,
                                &sA[buf][arow[cc] * BK + akp[cc]]);
    };
    auto loadB = [&](int k0) -> uint4 {
        uint4 v = make_uint4(0u, 0u, 0u, 0u);
        if (bvalid) v = *(const uint4*)(B + (size_t)(k0 + bk) * ldb + bn + bn8);
        return v;
    };
    auto storeB = [&](int buf, uint4 v) {
        unsigned u[4] = {v.x, v.y, v.z, v.w};
#pragma unroll
        for (int j = 0; j < 8; ++j) {
            unsigned short h = (unsigned short)(u[j >> 1] >> ((j & 1) * 16));
            sB[buf][(bn8 + j) * BK + bk] = h;               // transpose to [n][k]
        }
    };

    v8f zero = {0.f, 0.f, 0.f, 0.f, 0.f, 0.f, 0.f, 0.f};
    v8f acc[2][2];
    acc[0][0] = zero; acc[0][1] = zero; acc[1][0] = zero; acc[1][1] = zero;

    union Frag { v16bf v; unsigned int u[8]; };

    const int nk = K / BK;

    // prologue: stage tile 0 into buffer 0
    {
        uint4 v0 = loadB(0);
        issueA(0, 0);
        storeB(0, v0);
        wait_async0();
    }
    __syncthreads();

    for (int i = 0; i < nk; ++i) {
        const int cur = i & 1;
        const int nxt = cur ^ 1;
        const bool has_next = (i + 1) < nk;

        uint4 vBn = make_uint4(0u, 0u, 0u, 0u);
        if (has_next) {
            vBn = loadB((i + 1) * BK);        // overlap with compute below
            issueA((i + 1) * BK, nxt);        // async copy overlaps WMMA
        }

        // build fragments per ISA 7.12.2 layouts and multiply-accumulate
        Frag a[2], b[2];
#pragma unroll
        for (int t = 0; t < 2; ++t) {
            const unsigned int* ar = (const unsigned int*)(&sA[cur][(size_t)(waveM + t * 16 + l16) * BK]);
            const unsigned int* br = (const unsigned int*)(&sB[cur][(size_t)(waveN + t * 16 + l16) * BK]);
#pragma unroll
            for (int p = 0; p < 8; ++p) {
                // A: K pairs {0..7,16..23} (lanes 0-15) / {8..15,24..31} (lanes 16-31)
                int k2a = ((p < 4) ? p : (p + 4)) + half * 4;
                // B: K pairs {0..7,16..23} (lanes 0-15) / {8..15,24..31} (lanes 16-31)
                int k2b = ((p < 4) ? p : (p + 4)) + half * 4;
                a[t].u[p] = ar[k2a];
                b[t].u[p] = br[k2b];
            }
        }
#pragma unroll
        for (int mt = 0; mt < 2; ++mt)
#pragma unroll
            for (int nt = 0; nt < 2; ++nt)
                acc[mt][nt] = __builtin_amdgcn_wmma_f32_16x16x32_bf16(
                    false, a[mt].v, false, b[nt].v, (short)0, acc[mt][nt], false, false);

        if (has_next) {
            storeB(nxt, vBn);
            wait_async0();                    // this wave's async copies landed
        }
        __syncthreads();                      // everyone done reading cur / writing nxt
    }

    // store: C VGPR i -> M = i + half*8, N = l16
#pragma unroll
    for (int mt = 0; mt < 2; ++mt)
#pragma unroll
        for (int nt = 0; nt < 2; ++nt)
#pragma unroll
            for (int i = 0; i < 8; ++i) {
                int gr = bm + waveM + mt * 16 + half * 8 + i;
                int gc = bn + waveN + nt * 16 + l16;
                if (gr < M && gc < Ncols) {
                    float v = acc[mt][nt][i];
                    if (bias) v += bias[gc];
                    C[(size_t)gr * ldc + gc] = v;
                }
            }
}

// ---------------- log-softmax over 40 classes, one wave per row, in place ----------------
__global__ __launch_bounds__(256) void k_log_softmax40(float* __restrict__ z, int rows) {
    int row = blockIdx.x * 8 + (threadIdx.x >> 5);
    if (row >= rows) return;
    int lane = threadIdx.x & 31;
    float* p = z + (size_t)row * OUTD;
    float a = p[lane];
    float b = (lane < OUTD - 32) ? p[32 + lane] : -3.4e38f;
    float m = fmaxf(a, b);
#pragma unroll
    for (int off = 16; off; off >>= 1) m = fmaxf(m, __shfl_xor(m, off, 32));
    float s = __expf(a - m) + ((lane < OUTD - 32) ? __expf(b - m) : 0.f);
#pragma unroll
    for (int off = 16; off; off >>= 1) s += __shfl_xor(s, off, 32);
    float ls = __logf(s);
    p[lane] = a - m - ls;
    if (lane < OUTD - 32) p[32 + lane] = b - m - ls;
}

// ---------------- host-side launcher ----------------
extern "C" void kernel_launch(void* const* d_in, const int* in_sizes, int n_in,
                              void* d_out, int out_size, void* d_ws, size_t ws_size,
                              hipStream_t stream) {
    (void)in_sizes; (void)n_in; (void)out_size; (void)ws_size;

    const float* x      = (const float*)d_in[0];
    const int*   ei     = (const int*)d_in[1];     // [2][E]
    const float* W_in   = (const float*)d_in[2];
    const float* b_in   = (const float*)d_in[3];
    const float* W_hid  = (const float*)d_in[4];   // [3][256][256]
    const float* b_hid  = (const float*)d_in[5];   // [3][256]
    const float* gamma  = (const float*)d_in[6];   // [4][256]
    const float* beta   = (const float*)d_in[7];
    const float* W_out  = (const float*)d_in[8];   // [1024][40]
    const float* b_out  = (const float*)d_in[9];
    float* zout = (float*)d_out;                   // [N][40]

    const int* srcI = ei;
    const int* dstI = ei + EE;

    // ---- workspace carve-up ----
    char* ws = (char*)d_ws;
    size_t off = 0;
    auto carve = [&](size_t bytes) -> char* {
        char* p = ws + off;
        off += (bytes + 255) & ~(size_t)255;
        return p;
    };
    unsigned short* x_bf   = (unsigned short*)carve((size_t)NN * IND * 2);
    unsigned short* concat = (unsigned short*)carve((size_t)NN * LL * HD * 2);  // JK cat, bf16
    float*          hW     = (float*)carve((size_t)NN * HD * 4);
    float*          agg    = (float*)carve((size_t)NN * HD * 4);
    float*          deg    = (float*)carve((size_t)NN * 4);
    float*          dinv   = (float*)carve((size_t)NN * 4);
    float*          invd   = (float*)carve((size_t)NN * 4);
    float*          bstats = (float*)carve(512 * 4);        // bsum | bsq adjacent
    unsigned short* wb_in  = (unsigned short*)carve((size_t)IND * HD * 2);
    unsigned short* wb_hid = (unsigned short*)carve((size_t)3 * HD * HD * 2);
    unsigned short* wb_out = (unsigned short*)carve((size_t)LL * HD * OUTD * 2);
    float* bsum = bstats, *bsq = bstats + 256;

    auto cdiv = [](long long a, long long b) { return (int)((a + b - 1) / b); };

    // ---- bf16 conversions ----
    k_f32_to_bf16<<<cdiv((long long)NN * IND, 256), 256, 0, stream>>>(x, x_bf, (long long)NN * IND);
    k_f32_to_bf16<<<cdiv(IND * HD, 256), 256, 0, stream>>>(W_in, wb_in, IND * HD);
    k_f32_to_bf16<<<cdiv(3 * HD * HD, 256), 256, 0, stream>>>(W_hid, wb_hid, 3 * HD * HD);
    k_f32_to_bf16<<<cdiv(LL * HD * OUTD, 256), 256, 0, stream>>>(W_out, wb_out, LL * HD * OUTD);

    // ---- degree / gcn norms ----
    k_zero_f32<<<cdiv(NN, 256), 256, 0, stream>>>(deg, NN);
    k_edge_deg<<<cdiv(EE, 256), 256, 0, stream>>>(dstI, deg, EE);
    k_node_norm<<<cdiv(NN, 256), 256, 0, stream>>>(deg, dinv, invd, NN);

    const int gridM = cdiv(NN, BM);   // 1329

    for (int layer = 0; layer < LL; ++layer) {
        // 1) dense projection h @ W  (WMMA, async-LDS double buffered)
        const unsigned short* Ain = (layer == 0) ? x_bf : (concat + (size_t)(layer - 1) * HD);
        int lda = (layer == 0) ? IND : (LL * HD);
        int K   = (layer == 0) ? IND : HD;
        const unsigned short* Bw = (layer == 0) ? wb_in : (wb_hid + (size_t)(layer - 1) * HD * HD);
        dim3 gg(gridM, cdiv(HD, BN));
        k_gemm_bf16_wmma<<<gg, 256, 0, stream>>>(Ain, lda, Bw, HD, nullptr, hW, HD, NN, K, HD);

        // 2) sparse aggregation with symmetric norm
        k_zero_f32<<<cdiv((long long)NN * HD, 256), 256, 0, stream>>>(agg, (long long)NN * HD);
        k_aggregate<<<cdiv(EE, 8), 256, 0, stream>>>(hW, srcI, dstI, dinv, agg, EE);

        // 3) self-loop + bias + batch statistics
        const float* bl = (layer == 0) ? b_in : (b_hid + (size_t)(layer - 1) * HD);
        k_zero_f32<<<2, 256, 0, stream>>>(bstats, 512);
        k_finish_stats<<<cdiv(NN, 64), 256, 0, stream>>>(agg, hW, invd, bl, bsum, bsq, NN);

        // 4) BN + ReLU + pack bf16 into JK concat slice
        k_bn_relu_pack<<<cdiv((long long)NN * HD, 256), 256, 0, stream>>>(
            agg, bsum, bsq, gamma + (size_t)layer * HD, beta + (size_t)layer * HD,
            concat + (size_t)layer * HD, NN, 1.0f / (float)NN);
    }

    // ---- JK 'cat' output projection:  [N,1024] @ [1024,40] + b_out  (WMMA) ----
    dim3 go(gridM, cdiv(OUTD, BN));
    k_gemm_bf16_wmma<<<go, 256, 0, stream>>>(concat, LL * HD, wb_out, OUTD, b_out,
                                             zout, OUTD, NN, LL * HD, OUTD);

    // ---- log-softmax in place ----
    k_log_softmax40<<<cdiv(NN, 8), 256, 0, stream>>>(zout, NN);
}
